// LongcatMoe_60129542614
// MI455X (gfx1250) — matrix operand order repaired and checked by reading, not compiled, not run on hardware
//
#include <hip/hip_runtime.h>
#include <hip/hip_bf16.h>

// ---------------------------------------------------------------------------
// LongCat MoE for MI455X (gfx1250, wave32, WMMA bf16 16x16x32).
// Weight traffic (768MB fp32) dominates -> stream weights once per 32-token
// tile per expert; convert with v_cvt_pk_bf16_f32 (1 VALU per 2 elems);
// stage activations via global_load_async_to_lds_b128 (ASYNCcnt path).
// ---------------------------------------------------------------------------

#define TOP_K        4
#define N_ROUTED     32
#define E_TOT        40          // routed + zero experts
#define HIDDEN       2048
#define INTER        1024
#define TOKENS       1024
#define RSF          1.0f
#define M_TILE       32          // tokens per FFN block

typedef __attribute__((ext_vector_type(16))) __bf16        v16bf;
typedef __attribute__((ext_vector_type(8)))  float         v8f;
typedef __attribute__((ext_vector_type(4)))  unsigned int  v4u;

// fp32 -> bf16 RNE, scalar fallback (used on cold paths only)
__device__ __forceinline__ unsigned short f32_bf16(float f) {
    union { float f; unsigned u; } v; v.f = f;
    unsigned r = v.u + 0x7FFFu + ((v.u >> 16) & 1u);
    return (unsigned short)(r >> 16);
}

// Packed fp32x2 -> bf16x2 in one VALU op (hot path).
__device__ __forceinline__ unsigned pk_bf16(float a, float b) {
    unsigned r;
    asm("v_cvt_pk_bf16_f32 %0, %1, %2" : "=v"(r) : "v"(a), "v"(b));
    return r;
}

// A-matrix fragment (16x32 bf16) from LDS, row-major [rows][stride] bf16.
// Layout used here (contiguous k, same as B): lane<16: M=lane, K=k0+0..15;
//             lane>=16: M=lane-16, K=k0+16..31.
__device__ __forceinline__ v16bf lds_a_frag(const unsigned short* base,
                                            int m0, int stride, int k0) {
    int lane = threadIdx.x & 31;
    int m  = m0 + (lane & 15);
    int kb = (lane >> 4) << 4;                 // 0 or 16
    const unsigned short* p = base + m * stride + k0 + kb;
    union { v4u q[2]; v16bf v; } c;
    c.q[0] = *(const v4u*)(p);                 // K = k0+kb .. +7   (ds_load_b128)
    c.q[1] = *(const v4u*)(p + 8);             // K = k0+kb+8 .. +15
    return c.v;
}

// B-matrix fragment (32x16 bf16) from a global fp32 weight matrix stored
// row-major [N][ldk] (so B[k,n] = W[n,k] is contiguous along k).
// Layout used here (contiguous k, same as A): lane holds column n = n0+(lane&15), K = kb..kb+15, kb = 16*(lane>=16).
__device__ __forceinline__ v16bf glb_b_frag(const float* __restrict__ wbase,
                                            int n0, size_t ldk, int k0) {
    int lane = threadIdx.x & 31;
    int n  = n0 + (lane & 15);
    int kb = (lane >> 4) << 4;                 // 0 or 16
    const float* p = wbase + (size_t)n * ldk + (size_t)(k0 + kb);
    __builtin_prefetch(p + 64, 0, 1);          // global_prefetch_b8 on weight stream
    const float4* q = (const float4*)p;        // 64B aligned
    float4 f0 = q[0], f1 = q[1], f2 = q[2], f3 = q[3];
    union { unsigned u[8]; v16bf v; } c;
    c.u[0] = pk_bf16(f0.x, f0.y); c.u[1] = pk_bf16(f0.z, f0.w);
    c.u[2] = pk_bf16(f1.x, f1.y); c.u[3] = pk_bf16(f1.z, f1.w);
    c.u[4] = pk_bf16(f2.x, f2.y); c.u[5] = pk_bf16(f2.z, f2.w);
    c.u[6] = pk_bf16(f3.x, f3.y); c.u[7] = pk_bf16(f3.z, f3.w);
    return c.v;
}

// ---------------------------------------------------------------------------
// Kernel 1: router logits -> sigmoid -> top-4 (on score+bias) -> weights.
// ---------------------------------------------------------------------------
__global__ void router_topk_kernel(const float* __restrict__ x,
                                   const float* __restrict__ rw,
                                   const float* __restrict__ bias,
                                   int*   __restrict__ topk_id,
                                   float* __restrict__ topk_w,
                                   float* __restrict__ zero_w) {
    __shared__ float xr[HIDDEN];
    __shared__ float sc[E_TOT];
    const int t   = blockIdx.x;
    const int tid = threadIdx.x;                        // 64 threads

    const float4* xv  = (const float4*)(x + (size_t)t * HIDDEN);
    float4*       xrv = (float4*)xr;
    for (int i = tid; i < HIDDEN / 4; i += 64) xrv[i] = xv[i];
    __syncthreads();

    if (tid < E_TOT) {
        const float* wr = rw + (size_t)tid * HIDDEN;
        float acc = 0.f;
        for (int h = 0; h < HIDDEN; ++h) acc = fmaf(xr[h], wr[h], acc);
        sc[tid] = 1.f / (1.f + expf(-acc));             // sigmoid score
    }
    __syncthreads();

    if (tid == 0) {
        int   ids[TOP_K];
        float vals[TOP_K];
        #pragma unroll
        for (int k = 0; k < TOP_K; ++k) { ids[k] = 0; vals[k] = -1e30f; }
        for (int e = 0; e < E_TOT; ++e) {
            float v = sc[e] + bias[e];                  // selection key
            int k = TOP_K;
            while (k > 0 && v > vals[k - 1]) k--;       // ties keep earlier index
            if (k < TOP_K) {
                for (int j = TOP_K - 1; j > k; --j) { vals[j] = vals[j-1]; ids[j] = ids[j-1]; }
                vals[k] = v; ids[k] = e;
            }
        }
        float zw = 0.f;
        #pragma unroll
        for (int k = 0; k < TOP_K; ++k) {
            int e = ids[k];
            float w = sc[e];                            // raw sigmoid score (no renorm)
            topk_id[t * TOP_K + k] = e;
            topk_w [t * TOP_K + k] = w;
            if (e >= N_ROUTED) zw += w;                 // zero (identity) experts
        }
        zero_w[t] = zw;
    }
}

// ---------------------------------------------------------------------------
// Kernel 2/3: per-expert token bucketing.
// ---------------------------------------------------------------------------
__global__ void zero_counts_kernel(int* __restrict__ cnt) {
    if (threadIdx.x < N_ROUTED) cnt[threadIdx.x] = 0;
}

__global__ void scatter_kernel(const int*   __restrict__ topk_id,
                               const float* __restrict__ topk_w,
                               int*   __restrict__ cnt,
                               int*   __restrict__ btok,
                               float* __restrict__ bw) {
    int t = blockIdx.x * blockDim.x + threadIdx.x;
    if (t >= TOKENS) return;
    #pragma unroll
    for (int k = 0; k < TOP_K; ++k) {
        int e = topk_id[t * TOP_K + k];
        if (e < N_ROUTED) {
            int pos = atomicAdd(&cnt[e], 1);
            btok[e * TOKENS + pos] = t;
            bw  [e * TOKENS + pos] = topk_w[t * TOP_K + k] * RSF;
        }
    }
}

// ---------------------------------------------------------------------------
// Kernel 4: out = x * zero_weight * RSF  (also overwrites poisoned d_out).
// ---------------------------------------------------------------------------
__global__ void init_out_kernel(const float* __restrict__ x,
                                const float* __restrict__ zero_w,
                                float* __restrict__ out) {
    int i = blockIdx.x * blockDim.x + threadIdx.x;      // float4 index over T*H/4
    int t = i >> 9;                                     // (i*4)/HIDDEN
    float w = zero_w[t] * RSF;
    float4 v = ((const float4*)x)[i];
    v.x *= w; v.y *= w; v.z *= w; v.w *= w;
    ((float4*)out)[i] = v;
}

// ---------------------------------------------------------------------------
// Kernel 4b: pre-convert activations fp32 -> bf16 (read once, 8MB -> 4MB),
// so the FFN kernel can async-DMA them straight into LDS with no VALU.
// ---------------------------------------------------------------------------
__global__ void x_to_bf16_kernel(const float* __restrict__ x,
                                 unsigned* __restrict__ xbf_packed) {
    int i = blockIdx.x * blockDim.x + threadIdx.x;      // float4 index over T*H/4
    float4 f = ((const float4*)x)[i];
    uint2 o;
    o.x = pk_bf16(f.x, f.y);
    o.y = pk_bf16(f.z, f.w);
    ((uint2*)xbf_packed)[i] = o;
}

// ---------------------------------------------------------------------------
// Kernel 5: expert SwiGLU FFN on gathered 32-token tiles, bf16 WMMA.
// grid = (TOKENS/M_TILE, N_ROUTED), 256 threads (8 wave32).
// Dynamic LDS: xs[32][2048] bf16 (128KB) + hs[32][1024] bf16 (64KB) + meta.
// ---------------------------------------------------------------------------
__global__ void moe_ffn_kernel(const unsigned short* __restrict__ xbf,
                               const float* __restrict__ w1g,
                               const float* __restrict__ w1u,
                               const float* __restrict__ w2,
                               const int*   __restrict__ cnt,
                               const int*   __restrict__ btok,
                               const float* __restrict__ bw,
                               float* __restrict__ out) {
    extern __shared__ unsigned char smem[];
    unsigned short* xs = (unsigned short*)smem;                              // [M_TILE][HIDDEN]
    unsigned short* hs = (unsigned short*)(smem + (size_t)M_TILE * HIDDEN * 2); // [M_TILE][INTER]
    int*   ti = (int*)  (smem + (size_t)M_TILE * (HIDDEN + INTER) * 2);
    float* wt = (float*)(ti + M_TILE);

    const int e     = blockIdx.y;
    const int start = blockIdx.x * M_TILE;
    const int count = cnt[e];
    if (start >= count) return;                          // block-uniform exit

    const int tid = threadIdx.x;

    if (tid < M_TILE) {
        int idx = start + tid;
        ti[tid] = (idx < count) ? btok[e * TOKENS + idx] : 0;
        wt[tid] = (idx < count) ? bw  [e * TOKENS + idx] : 0.f;  // pad rows weigh 0
    }
    __syncthreads();

    // Stage gathered bf16 activation rows into LDS with async copies:
    // 8192 x 16B chunks (32 rows x 256 chunks), 32 per thread, ASYNCcnt-tracked.
    {
        unsigned xs_base = (unsigned)(uintptr_t)xs;      // LDS offset (low 32b of generic addr)
        for (int i = tid; i < M_TILE * (HIDDEN / 8); i += 256) {
            int row = i >> 8;                            // / (HIDDEN/8)
            int c   = i & 255;                           // 16B chunk in row
            const unsigned short* g = xbf + (size_t)ti[row] * HIDDEN + c * 8;
            unsigned l = xs_base + (unsigned)(row * (HIDDEN * 2) + c * 16);
            asm volatile("global_load_async_to_lds_b128 %0, %1, off"
                         :: "v"(l), "v"(g) : "memory");
        }
        asm volatile("s_wait_asynccnt 0x0" ::: "memory");
    }
    __syncthreads();

    const int wave = tid >> 5;
    const int lane = tid & 31;
    const float* w1g_e = w1g + (size_t)e * INTER * HIDDEN;
    const float* w1u_e = w1u + (size_t)e * INTER * HIDDEN;
    const float* w2_e  = w2  + (size_t)e * HIDDEN * INTER;

    // ---- Phase 1: g,u = X @ W1g^T, X @ W1u^T ; h = silu(g)*u -> hs ----
    for (int wtile = wave; wtile < (INTER / 16) * 2; wtile += 8) {
        const int n0 = (wtile >> 1) * 16;                // inter column tile
        const int m0 = (wtile & 1) * 16;                 // token half-tile
        v8f ag = {}, au = {};
        for (int k0 = 0; k0 < HIDDEN; k0 += 32) {
            v16bf a  = lds_a_frag(xs, m0, HIDDEN, k0);
            v16bf bg = glb_b_frag(w1g_e, n0, HIDDEN, k0);
            v16bf bu = glb_b_frag(w1u_e, n0, HIDDEN, k0);
            ag = __builtin_amdgcn_wmma_f32_16x16x32_bf16(false, a, false, bg,
                                                         (short)0, ag, false, false);
            au = __builtin_amdgcn_wmma_f32_16x16x32_bf16(false, a, false, bu,
                                                         (short)0, au, false, false);
        }
        const int n  = n0 + (lane & 15);
        const int mb = m0 + ((lane >> 4) << 3);          // C layout: VGPR r -> M = mb+r
        #pragma unroll
        for (int r = 0; r < 8; ++r) {
            float g = ag[r], u = au[r];
            float h = (g / (1.f + expf(-g))) * u;        // silu(g) * u
            hs[(mb + r) * INTER + n] = f32_bf16(h);
        }
    }
    __syncthreads();

    // ---- Phase 2: Y = H @ W2^T, weighted atomic combine into out ----
    for (int wtile = wave; wtile < (HIDDEN / 16) * 2; wtile += 8) {
        const int n0 = (wtile >> 1) * 16;                // hidden column tile
        const int m0 = (wtile & 1) * 16;
        v8f acc = {};
        for (int k0 = 0; k0 < INTER; k0 += 32) {
            v16bf a = lds_a_frag(hs, m0, INTER, k0);
            v16bf b = glb_b_frag(w2_e, n0, INTER, k0);
            acc = __builtin_amdgcn_wmma_f32_16x16x32_bf16(false, a, false, b,
                                                          (short)0, acc, false, false);
        }
        const int n  = n0 + (lane & 15);
        const int mb = m0 + ((lane >> 4) << 3);
        #pragma unroll
        for (int r = 0; r < 8; ++r) {
            int m = mb + r;
            float v = acc[r] * wt[m];
            atomicAdd(&out[(size_t)ti[m] * HIDDEN + n], v);  // global_atomic_add_f32
        }
    }
}

// ---------------------------------------------------------------------------
// Host-side launcher. Workspace layout (offsets 128B-aligned):
//   cnt[32] | topk_id[T*4] | topk_w[T*4] | zero_w[T] | btok[32*T] | bw[32*T]
//   | xbf[T*H] bf16 (4MB)
// ---------------------------------------------------------------------------
extern "C" void kernel_launch(void* const* d_in, const int* in_sizes, int n_in,
                              void* d_out, int out_size, void* d_ws, size_t ws_size,
                              hipStream_t stream) {
    const float* x    = (const float*)d_in[0];
    const float* rw   = (const float*)d_in[1];
    const float* bias = (const float*)d_in[2];
    const float* w1g  = (const float*)d_in[3];
    const float* w1u  = (const float*)d_in[4];
    const float* w2   = (const float*)d_in[5];
    float* out = (float*)d_out;

    unsigned char* ws = (unsigned char*)d_ws;
    size_t o = 0;
    int*   cnt     = (int*)  (ws + o); o += 128;
    int*   topk_id = (int*)  (ws + o); o += (size_t)TOKENS * TOP_K * sizeof(int);
    float* topk_w  = (float*)(ws + o); o += (size_t)TOKENS * TOP_K * sizeof(float);
    float* zero_w  = (float*)(ws + o); o += (size_t)TOKENS * sizeof(float);
    int*   btok    = (int*)  (ws + o); o += (size_t)N_ROUTED * TOKENS * sizeof(int);
    float* bw      = (float*)(ws + o); o += (size_t)N_ROUTED * TOKENS * sizeof(float);
    unsigned short* xbf = (unsigned short*)(ws + o);    // TOKENS*HIDDEN bf16

    router_topk_kernel<<<TOKENS, 64, 0, stream>>>(x, rw, bias, topk_id, topk_w, zero_w);
    zero_counts_kernel<<<1, 32, 0, stream>>>(cnt);
    scatter_kernel<<<TOKENS / 256, 256, 0, stream>>>(topk_id, topk_w, cnt, btok, bw);
    init_out_kernel<<<(TOKENS * HIDDEN / 4) / 256, 256, 0, stream>>>(x, zero_w, out);
    x_to_bf16_kernel<<<(TOKENS * HIDDEN / 4) / 256, 256, 0, stream>>>(x, (unsigned*)xbf);

    size_t lds = (size_t)M_TILE * (HIDDEN + INTER) * 2   // xs + hs (bf16)
               + (size_t)M_TILE * (sizeof(int) + sizeof(float));
    moe_ffn_kernel<<<dim3(TOKENS / M_TILE, N_ROUTED), 256, lds, stream>>>(
        xbf, w1g, w1u, w2, cnt, btok, bw, out);
}
